// DecoderLayer_79070347919388
// MI455X (gfx1250) — matrix operand order repaired and checked by reading, not compiled, not run on hardware
//
#include <hip/hip_runtime.h>
#include <hip/hip_bf16.h>
#include <stdint.h>

// Problem constants (match reference)
#define Bd   2
#define Sd   2048
#define Dd   1024
#define Hd   16
#define DHd  64
#define DFFd 4096
#define EPSd 1e-6f
#define NEGd (-1e9f)

typedef __attribute__((ext_vector_type(16))) __bf16 v16bf;
typedef __attribute__((ext_vector_type(8)))  float  v8f;

union FragBF { v16bf v; uint4 u[2]; };

__device__ __forceinline__ unsigned short f2bf(float f) {
  unsigned int u = __float_as_uint(f);
  u += 0x7fffu + ((u >> 16) & 1u);   // round-to-nearest-even truncation
  return (unsigned short)(u >> 16);
}

// CDNA5 async Global->LDS copy (16B per lane), tracked by ASYNCcnt.
// ldsOff must be derived from the shared object's address (ptrtoint) so the
// object escapes and the "memory" clobber covers the LDS writes.
__device__ __forceinline__ void async_g2l_b128(unsigned int ldsOff, const void* g) {
  unsigned long long ga = (unsigned long long)(uintptr_t)g;
  asm volatile("global_load_async_to_lds_b128 %0, %1, off"
               :: "v"(ldsOff), "v"(ga) : "memory");
}
__device__ __forceinline__ void wait_asynccnt0() {
  asm volatile("s_wait_asynccnt 0" ::: "memory");
}
__device__ __forceinline__ void wait_asynccnt3() {   // oldest tile landed
  asm volatile("s_wait_asynccnt 3" ::: "memory");
}

// ---------------------------------------------------------------------------
// Uniform bf16 GEMM:  C = alpha * (A x B^T) + bias  [+ReLU]
//   A : bf16, row-major [m][k], leading dim lda
//   B : bf16, stored [n][k] (weights pre-transposed; k-matrix & v^T naturally)
// Double-buffered async Global->LDS staging (ASYNCcnt pipeline), ds_load_b128
// fragment gathers per the ISA 16-bit A/B VGPR striping, MACs on
// v_wmma_f32_16x16x32_bf16.  Epilogue can emit f32 C, bf16 C, and/or the
// per-head transposed bf16 copy vt[(b*H+h)*DH+dh][s] (V projection).
// ---------------------------------------------------------------------------
struct GemmArgs {
  const unsigned short* A; const unsigned short* B; const float* bias;
  float* Cf; unsigned short* Ch; unsigned short* Vt;
  long long lda, ldb, ldc;
  long long sAb, sAh, sBb, sBh, sCb, sCh;
  int Hq, M, N, K;
  float alpha; int relu;
};

#define TM 128
#define TN 64
#define TK 32
#define LSA 48                        // LDS row stride in ushorts (32 data + pad)
#define BUF_USH ((TM + TN) * LSA)     // ushorts per staging buffer (9216 = 18KB)

__global__ __launch_bounds__(256)
void wmma_gemm_kernel(GemmArgs g) {
  __shared__ __align__(16) unsigned short smem[2 * BUF_USH];

  const int tid  = threadIdx.x;
  const int lane = tid & 31;
  const int wave = tid >> 5;
  const int wm   = wave & 3;   // 4 waves along M -> 32 rows each
  const int wn   = wave >> 2;  // 2 waves along N -> 32 cols each

  const int z  = blockIdx.z;
  const int bb = z / g.Hq;
  const int hh = z % g.Hq;
  const unsigned short* Ab = g.A + (long long)bb * g.sAb + (long long)hh * g.sAh;
  const unsigned short* Bb = g.B + (long long)bb * g.sBb + (long long)hh * g.sBh;

  const int m0 = blockIdx.y * TM;
  const int n0 = blockIdx.x * TN;

  // Static staging assignment (per thread): A = 2 x 16B chunks, B = 1 x 16B.
  const int ca0_row = tid >> 2,          ca_ch = tid & 3;    // A rows 0..63
  const int ca1_row = (tid + 256) >> 2;                      // A rows 64..127
  const int cb_row  = tid >> 2;                              // B rows 0..63

  // LDS offsets derived from the object's address: low 32 bits of the flat
  // address are the allocation-relative dsaddr (ISA 10.2), and the ptrtoint
  // escape keeps the ds_load gathers from being folded away.
  const unsigned ldsBase = (unsigned)(uintptr_t)(&smem[0]);
  unsigned lA0o[2], lA1o[2], lBo[2];
#pragma unroll
  for (int b = 0; b < 2; ++b) {
    const unsigned base = ldsBase + (unsigned)(b * BUF_USH * 2);
    lA0o[b] = base + (unsigned)((ca0_row * LSA + ca_ch * 8) * 2);
    lA1o[b] = base + (unsigned)((ca1_row * LSA + ca_ch * 8) * 2);
    lBo[b]  = base + (unsigned)(((TM + cb_row) * LSA + ca_ch * 8) * 2);
  }
  const unsigned short* gA0 = Ab + (long long)(m0 + ca0_row) * g.lda + ca_ch * 8;
  const unsigned short* gA1 = Ab + (long long)(m0 + ca1_row) * g.lda + ca_ch * 8;
  const unsigned short* gB  = Bb + (long long)(n0 + cb_row)  * g.ldb + ca_ch * 8;

  v8f acc00 = {}; v8f acc01 = {}; v8f acc10 = {}; v8f acc11 = {};

  const int kTiles = g.K / TK;
  // prologue: stage tile 0 into buffer 0
  async_g2l_b128(lA0o[0], gA0);
  async_g2l_b128(lA1o[0], gA1);
  async_g2l_b128(lBo[0],  gB);

  for (int kt = 0; kt < kTiles; ++kt) {
    const int  cur  = kt & 1;
    const bool more = (kt + 1) < kTiles;
    if (more) {               // overlap next tile's DMA with this tile's math
      const int k1 = (kt + 1) * TK;
      async_g2l_b128(lA0o[cur ^ 1], gA0 + k1);
      async_g2l_b128(lA1o[cur ^ 1], gA1 + k1);
      async_g2l_b128(lBo[cur ^ 1],  gB  + k1);
      wait_asynccnt3();       // in-order: the 3 oldest (tile kt) have landed
    } else {
      wait_asynccnt0();
    }
    __syncthreads();          // tile kt visible from every wave

    // ---- gather WMMA fragments (ISA 16-bit A 16x32 / B 32x16 striping) ----
    const unsigned short* lA = smem + cur * BUF_USH;
    const unsigned short* lB = lA + TM * LSA;
    const int half = lane >> 4;   // lane group 0..15 / 16..31
    const int idx  = lane & 15;   // A: row m; B: col n
    FragBF a0, a1, b0, b1;
    {
      const unsigned short* p = &lA[(wm * 32 + idx) * LSA];
      a0.u[0] = *(const uint4*)(p + half * 8);        // K = half*8 + 0..7
      a0.u[1] = *(const uint4*)(p + 16 + half * 8);   // K = 16 + half*8 + 0..7
      const unsigned short* q = &lA[(wm * 32 + 16 + idx) * LSA];
      a1.u[0] = *(const uint4*)(q + half * 8);
      a1.u[1] = *(const uint4*)(q + 16 + half * 8);
    }
    {
      const unsigned short* p = &lB[(wn * 32 + idx) * LSA] + half * 8; // K = half*8 + 0..7, 16 + half*8 + 0..7
      b0.u[0] = *(const uint4*)(p);
      b0.u[1] = *(const uint4*)(p + 16);
      const unsigned short* q = &lB[(wn * 32 + 16 + idx) * LSA] + half * 8;
      b1.u[0] = *(const uint4*)(q);
      b1.u[1] = *(const uint4*)(q + 16);
    }
    acc00 = __builtin_amdgcn_wmma_f32_16x16x32_bf16(false, a0.v, false, b0.v, (short)0, acc00, false, false);
    acc01 = __builtin_amdgcn_wmma_f32_16x16x32_bf16(false, a0.v, false, b1.v, (short)0, acc01, false, false);
    acc10 = __builtin_amdgcn_wmma_f32_16x16x32_bf16(false, a1.v, false, b0.v, (short)0, acc10, false, false);
    acc11 = __builtin_amdgcn_wmma_f32_16x16x32_bf16(false, a1.v, false, b1.v, (short)0, acc11, false, false);

    __syncthreads();          // reads done before this buffer is re-staged
  }

  // ---- epilogue: C/D layout -> VGPR r: m = r + (lane/16)*8, n = lane%16 ----
  float*          Cf = g.Cf ? g.Cf + (long long)bb * g.sCb + (long long)hh * g.sCh : nullptr;
  unsigned short* Ch = g.Ch ? g.Ch + (long long)bb * g.sCb + (long long)hh * g.sCh : nullptr;
  const int half = lane >> 4;
  const int ncol = lane & 15;
  const v8f* accs[4] = { &acc00, &acc01, &acc10, &acc11 };
  const int  moff[4] = { 0, 0, 16, 16 };
  const int  noff[4] = { 0, 16, 0, 16 };
#pragma unroll
  for (int t = 0; t < 4; ++t) {
    const v8f a = *accs[t];
#pragma unroll
    for (int r = 0; r < 8; ++r) {
      const int m = m0 + wm * 32 + moff[t] + half * 8 + r;
      const int n = n0 + wn * 32 + noff[t] + ncol;
      float v = a[r] * g.alpha;
      if (g.bias) v += g.bias[n];
      if (g.relu) v = fmaxf(v, 0.0f);
      if (Cf) Cf[(long long)m * g.ldc + n] = v;
      if (Ch) Ch[(long long)m * g.ldc + n] = f2bf(v);
      if (g.Vt) {  // vt[((b*H + h)*DH + dh)][s]  (per-head transposed copy)
        const int b2 = m / Sd, sp = m % Sd, h2 = n / DHd, dh = n % DHd;
        g.Vt[(((long long)b2 * Hd + h2) * DHd + dh) * Sd + sp] = f2bf(v);
      }
    }
  }
}

// ---------------------------------------------------------------------------
// f32 -> bf16 elementwise (vectorized x4)
// ---------------------------------------------------------------------------
__global__ __launch_bounds__(256)
void cvt_bf16_kernel(const float* __restrict__ s, unsigned short* __restrict__ d) {
  const long long i = (long long)blockIdx.x * 256 + threadIdx.x;
  const float4 f = ((const float4*)s)[i];
  uint2 o;
  o.x = (unsigned)f2bf(f.x) | ((unsigned)f2bf(f.y) << 16);
  o.y = (unsigned)f2bf(f.z) | ((unsigned)f2bf(f.w) << 16);
  ((uint2*)d)[i] = o;
}

// ---------------------------------------------------------------------------
// W (f32, [K][N]) -> Wt (bf16, [N][K])   tiled transpose
// ---------------------------------------------------------------------------
__global__ __launch_bounds__(256)
void transpose_bf16_kernel(const float* __restrict__ W, unsigned short* __restrict__ Wt,
                           int K, int N) {
  __shared__ float t[32][33];
  const int k0 = blockIdx.y * 32, n0 = blockIdx.x * 32;
  const int tx = threadIdx.x & 31, ty = threadIdx.x >> 5;  // 32 x 8
#pragma unroll
  for (int i = 0; i < 32; i += 8)
    t[ty + i][tx] = W[(long long)(k0 + ty + i) * N + n0 + tx];
  __syncthreads();
#pragma unroll
  for (int i = 0; i < 32; i += 8)
    Wt[(long long)(n0 + ty + i) * K + k0 + tx] = f2bf(t[tx][ty + i]);
}

// ---------------------------------------------------------------------------
// In-place masked softmax over rows of length S; also writes a bf16 copy.
// maskMode 0: mask[(row % S) * S + c]       maskMode 1: mask[(row/(H*S)) * S + c]
// ---------------------------------------------------------------------------
__global__ __launch_bounds__(256)
void softmax_mask_kernel(float* t, unsigned short* tb, const float* mask, int maskMode) {
  const long long r = blockIdx.x;
  const int tid = threadIdx.x;
  float* row = t + r * (long long)Sd;
  unsigned short* rowb = tb + r * (long long)Sd;
  const float* mrow;
  if (maskMode == 0) mrow = mask + (long long)(r % Sd) * Sd;
  else               mrow = mask + (long long)(r / ((long long)Hd * Sd)) * Sd;

  float loc[8];
  float mx = -3.0e38f;
#pragma unroll
  for (int i = 0; i < 8; ++i) {
    const int c = tid + i * 256;
    const float v = row[c] + mrow[c] * NEGd;
    loc[i] = v;
    mx = fmaxf(mx, v);
  }
  for (int o = 16; o > 0; o >>= 1) mx = fmaxf(mx, __shfl_xor(mx, o, 32));
  __shared__ float redM[8];
  __shared__ float redS[8];
  if ((tid & 31) == 0) redM[tid >> 5] = mx;
  __syncthreads();
  float m2 = redM[0];
#pragma unroll
  for (int w = 1; w < 8; ++w) m2 = fmaxf(m2, redM[w]);

  float sum = 0.f;
#pragma unroll
  for (int i = 0; i < 8; ++i) { loc[i] = __expf(loc[i] - m2); sum += loc[i]; }
  for (int o = 16; o > 0; o >>= 1) sum += __shfl_xor(sum, o, 32);
  if ((tid & 31) == 0) redS[tid >> 5] = sum;
  __syncthreads();
  float s2 = 0.f;
#pragma unroll
  for (int w = 0; w < 8; ++w) s2 += redS[w];
  const float inv = 1.0f / s2;
#pragma unroll
  for (int i = 0; i < 8; ++i) {
    const float v = loc[i] * inv;
    row[tid + i * 256]  = v;
    rowb[tid + i * 256] = f2bf(v);
  }
}

// ---------------------------------------------------------------------------
// out = LayerNorm(a + res) * gamma + beta ; optional bf16 copy  (cols = D)
// ---------------------------------------------------------------------------
__device__ __forceinline__ float block_sum256(float v, float* red, int tid) {
  for (int o = 16; o > 0; o >>= 1) v += __shfl_xor(v, o, 32);
  if ((tid & 31) == 0) red[tid >> 5] = v;
  __syncthreads();
  float s = 0.f;
#pragma unroll
  for (int w = 0; w < 8; ++w) s += red[w];
  __syncthreads();
  return s;
}

__global__ __launch_bounds__(256)
void add_layernorm_kernel(const float* __restrict__ a, const float* __restrict__ res,
                          const float* __restrict__ gamma, const float* __restrict__ beta,
                          float* __restrict__ out, unsigned short* __restrict__ outb) {
  const long long r = blockIdx.x;
  const int tid = threadIdx.x;
  const float* ar = a   + r * (long long)Dd;
  const float* rr = res + r * (long long)Dd;
  __shared__ float red[8];

  float loc[4];
  float s = 0.f;
#pragma unroll
  for (int i = 0; i < 4; ++i) {
    const int c = tid + i * 256;
    loc[i] = ar[c] + rr[c];
    s += loc[i];
  }
  const float mean = block_sum256(s, red, tid) * (1.0f / Dd);
  float vs = 0.f;
#pragma unroll
  for (int i = 0; i < 4; ++i) { const float d = loc[i] - mean; vs += d * d; }
  const float var = block_sum256(vs, red, tid) * (1.0f / Dd);
  const float inv = rsqrtf(var + EPSd);
#pragma unroll
  for (int i = 0; i < 4; ++i) {
    const int c = tid + i * 256;
    const float v = (loc[i] - mean) * inv * gamma[c] + beta[c];
    out[r * (long long)Dd + c] = v;
    if (outb) outb[r * (long long)Dd + c] = f2bf(v);
  }
}

// ---------------------------------------------------------------------------
// Host-side orchestration
// ---------------------------------------------------------------------------
static void launch_gemm(hipStream_t stream,
                        const unsigned short* A, long long lda,
                        const unsigned short* B, long long ldb,
                        const float* bias,
                        float* Cf, unsigned short* Ch, unsigned short* Vt, long long ldc,
                        int M, int N, int K, float alpha, int relu,
                        int batch, int Hq,
                        long long sAb, long long sAh,
                        long long sBb, long long sBh,
                        long long sCb, long long sCh) {
  GemmArgs g;
  g.A = A; g.B = B; g.bias = bias; g.Cf = Cf; g.Ch = Ch; g.Vt = Vt;
  g.lda = lda; g.ldb = ldb; g.ldc = ldc;
  g.sAb = sAb; g.sAh = sAh; g.sBb = sBb; g.sBh = sBh; g.sCb = sCb; g.sCh = sCh;
  g.Hq = Hq; g.M = M; g.N = N; g.K = K; g.alpha = alpha; g.relu = relu;
  dim3 grid(N / TN, M / TM, batch);
  wmma_gemm_kernel<<<grid, dim3(256), 0, stream>>>(g);
}

static void launch_transpose(hipStream_t stream, const float* W, unsigned short* Wt,
                             int K, int N) {
  transpose_bf16_kernel<<<dim3(N / 32, K / 32), dim3(256), 0, stream>>>(W, Wt, K, N);
}

extern "C" void kernel_launch(void* const* d_in, const int* in_sizes, int n_in,
                              void* d_out, int out_size, void* d_ws, size_t ws_size,
                              hipStream_t stream) {
  (void)in_sizes; (void)n_in; (void)out_size; (void)ws_size;

  const float* x   = (const float*)d_in[0];
  const float* enc = (const float*)d_in[1];
  const float* lam = (const float*)d_in[2];   // (1,1,S,S)
  const float* pam = (const float*)d_in[3];   // (B,1,1,S)
  const float* m1_Wq = (const float*)d_in[4];  const float* m1_qb = (const float*)d_in[5];
  const float* m1_Wk = (const float*)d_in[6];  const float* m1_kb = (const float*)d_in[7];
  const float* m1_Wv = (const float*)d_in[8];  const float* m1_vb = (const float*)d_in[9];
  const float* m1_Wo = (const float*)d_in[10]; const float* m1_ob = (const float*)d_in[11];
  const float* m2_Wq = (const float*)d_in[12]; const float* m2_qb = (const float*)d_in[13];
  const float* m2_Wk = (const float*)d_in[14]; const float* m2_kb = (const float*)d_in[15];
  const float* m2_Wv = (const float*)d_in[16]; const float* m2_vb = (const float*)d_in[17];
  const float* m2_Wo = (const float*)d_in[18]; const float* m2_ob = (const float*)d_in[19];
  const float* fW1 = (const float*)d_in[20];   const float* fb1 = (const float*)d_in[21];
  const float* fW2 = (const float*)d_in[22];   const float* fb2 = (const float*)d_in[23];
  const float* ln1g = (const float*)d_in[24];  const float* ln1b = (const float*)d_in[25];
  const float* ln2g = (const float*)d_in[26];  const float* ln2b = (const float*)d_in[27];
  const float* ln3g = (const float*)d_in[28];  const float* ln3b = (const float*)d_in[29];

  const long long NT  = (long long)Bd * Sd * Dd;        // 4 Mi
  const long long NAW = (long long)Bd * Hd * Sd * Sd;   // 128 Mi
  const long long DD  = (long long)Dd * Dd;             // 1 Mi
  const long long DF  = (long long)Dd * DFFd;           // 4 Mi

  // d_out: [ out3 | aw1 | aw2 ]  (all f32)
  float* out3 = (float*)d_out;
  float* aw1  = out3 + NT;
  float* aw2  = aw1 + NAW;

  // ---- workspace layout: f32 region first, then bf16 (ushort) region ----
  float* projf = (float*)d_ws;          // NT
  float* out1f = projf + NT;            // NT
  float* out2f = out1f + NT;            // NT
  unsigned short* us = (unsigned short*)(out2f + NT);
  unsigned short* xb    = us;  us += NT;
  unsigned short* encb  = us;  us += NT;
  unsigned short* wt_q1 = us;  us += DD;
  unsigned short* wt_k1 = us;  us += DD;
  unsigned short* wt_v1 = us;  us += DD;
  unsigned short* wt_o1 = us;  us += DD;
  unsigned short* wt_q2 = us;  us += DD;
  unsigned short* wt_k2 = us;  us += DD;
  unsigned short* wt_v2 = us;  us += DD;
  unsigned short* wt_o2 = us;  us += DD;
  unsigned short* w1t   = us;  us += DF;
  unsigned short* w2t   = us;  us += DF;
  unsigned short* qb    = us;  us += NT;
  unsigned short* kb    = us;  us += NT;
  unsigned short* vt    = us;  us += NT;   // per-head transposed V
  unsigned short* ctxb  = us;  us += NT;
  unsigned short* out1b = us;  us += NT;
  unsigned short* out2b = us;  us += NT;
  unsigned short* ffnb  = us;  us += (long long)Bd * Sd * DFFd;
  unsigned short* awb   = us;  us += NAW;  // reused for both MHAs

  const int   MS    = Bd * Sd;          // 4096
  const float iSqDH = 0.125f;           // 1/sqrt(64)
  const long long SS = (long long)Sd * Sd;

  // ================= one-time bf16 / transpose pre-pass =================
  cvt_bf16_kernel<<<NT / 4 / 256, 256, 0, stream>>>(x, xb);
  cvt_bf16_kernel<<<NT / 4 / 256, 256, 0, stream>>>(enc, encb);
  launch_transpose(stream, m1_Wq, wt_q1, Dd, Dd);
  launch_transpose(stream, m1_Wk, wt_k1, Dd, Dd);
  launch_transpose(stream, m1_Wv, wt_v1, Dd, Dd);
  launch_transpose(stream, m1_Wo, wt_o1, Dd, Dd);
  launch_transpose(stream, m2_Wq, wt_q2, Dd, Dd);
  launch_transpose(stream, m2_Wk, wt_k2, Dd, Dd);
  launch_transpose(stream, m2_Wv, wt_v2, Dd, Dd);
  launch_transpose(stream, m2_Wo, wt_o2, Dd, Dd);
  launch_transpose(stream, fW1, w1t, Dd, DFFd);
  launch_transpose(stream, fW2, w2t, DFFd, Dd);

  // ================= MHA 1 (masked self-attention) =================
  launch_gemm(stream, xb, Dd, wt_q1, Dd, m1_qb, nullptr, qb, nullptr, Dd,
              MS, Dd, Dd, 1.f, 0, 1, 1, 0,0, 0,0, 0,0);
  launch_gemm(stream, xb, Dd, wt_k1, Dd, m1_kb, nullptr, kb, nullptr, Dd,
              MS, Dd, Dd, 1.f, 0, 1, 1, 0,0, 0,0, 0,0);
  launch_gemm(stream, xb, Dd, wt_v1, Dd, m1_vb, nullptr, nullptr, vt, Dd,
              MS, Dd, Dd, 1.f, 0, 1, 1, 0,0, 0,0, 0,0);
  launch_gemm(stream, qb, Dd, kb, Dd, nullptr, aw1, nullptr, nullptr, Sd,
              Sd, Sd, DHd, iSqDH, 0, Bd * Hd, Hd,
              (long long)Sd * Dd, DHd, (long long)Sd * Dd, DHd,
              (long long)Hd * SS, SS);
  softmax_mask_kernel<<<Bd * Hd * Sd, 256, 0, stream>>>(aw1, awb, lam, 0);
  launch_gemm(stream, awb, Sd, vt, Sd, nullptr, nullptr, ctxb, nullptr, Dd,
              Sd, DHd, Sd, 1.f, 0, Bd * Hd, Hd,
              (long long)Hd * SS, SS, (long long)Hd * DHd * Sd, (long long)DHd * Sd,
              (long long)Sd * Dd, DHd);
  launch_gemm(stream, ctxb, Dd, wt_o1, Dd, m1_ob, projf, nullptr, nullptr, Dd,
              MS, Dd, Dd, 1.f, 0, 1, 1, 0,0, 0,0, 0,0);
  add_layernorm_kernel<<<MS, 256, 0, stream>>>(projf, x, ln1g, ln1b, out1f, out1b);

  // ================= MHA 2 (cross-attention) =================
  launch_gemm(stream, out1b, Dd, wt_q2, Dd, m2_qb, nullptr, qb, nullptr, Dd,
              MS, Dd, Dd, 1.f, 0, 1, 1, 0,0, 0,0, 0,0);
  launch_gemm(stream, encb, Dd, wt_k2, Dd, m2_kb, nullptr, kb, nullptr, Dd,
              MS, Dd, Dd, 1.f, 0, 1, 1, 0,0, 0,0, 0,0);
  launch_gemm(stream, encb, Dd, wt_v2, Dd, m2_vb, nullptr, nullptr, vt, Dd,
              MS, Dd, Dd, 1.f, 0, 1, 1, 0,0, 0,0, 0,0);
  launch_gemm(stream, qb, Dd, kb, Dd, nullptr, aw2, nullptr, nullptr, Sd,
              Sd, Sd, DHd, iSqDH, 0, Bd * Hd, Hd,
              (long long)Sd * Dd, DHd, (long long)Sd * Dd, DHd,
              (long long)Hd * SS, SS);
  softmax_mask_kernel<<<Bd * Hd * Sd, 256, 0, stream>>>(aw2, awb, pam, 1);
  launch_gemm(stream, awb, Sd, vt, Sd, nullptr, nullptr, ctxb, nullptr, Dd,
              Sd, DHd, Sd, 1.f, 0, Bd * Hd, Hd,
              (long long)Hd * SS, SS, (long long)Hd * DHd * Sd, (long long)DHd * Sd,
              (long long)Sd * Dd, DHd);
  launch_gemm(stream, ctxb, Dd, wt_o2, Dd, m2_ob, projf, nullptr, nullptr, Dd,
              MS, Dd, Dd, 1.f, 0, 1, 1, 0,0, 0,0, 0,0);
  add_layernorm_kernel<<<MS, 256, 0, stream>>>(projf, out1f, ln2g, ln2b, out2f, out2b);

  // ================= FFN =================
  launch_gemm(stream, out2b, Dd, w1t, Dd, fb1, nullptr, ffnb, nullptr, DFFd,
              MS, DFFd, Dd, 1.f, 1, 1, 1, 0,0, 0,0, 0,0);
  launch_gemm(stream, ffnb, DFFd, w2t, DFFd, fb2, projf, nullptr, nullptr, Dd,
              MS, Dd, DFFd, 1.f, 0, 1, 1, 0,0, 0,0, 0,0);
  add_layernorm_kernel<<<MS, 256, 0, stream>>>(projf, out2f, ln3g, ln3b, out3, nullptr);
}
